// PreEncodedGCN_22290880266881
// MI455X (gfx1250) — matrix operand order repaired and checked by reading, not compiled, not run on hardware
//
#include <hip/hip_runtime.h>
#include <hip/hip_bf16.h>
#include <stdint.h>

// ---------------------------------------------------------------------------
// PreEncodedGCN for gfx1250 (MI455X): bf16 WMMA GEMMs + fp32 atomic scatters.
// ---------------------------------------------------------------------------

#define C_DIM 256
#define NBAS  8
#define NREL  16
#define YW    2304   // NBAS*C + C (bases + root, fused into one GEMM)

typedef __attribute__((ext_vector_type(16))) __bf16 v16bf;
typedef __attribute__((ext_vector_type(8)))  float  v8f;

__device__ __forceinline__ unsigned short f2bf(float f) {
  unsigned int u = __float_as_uint(f);
  u += 0x7FFFu + ((u >> 16) & 1u);          // round-to-nearest-even
  return (unsigned short)(u >> 16);
}
__device__ __forceinline__ float bf2f(unsigned int h) {
  return __uint_as_float(h << 16);
}
__device__ __forceinline__ void unpack8(uint4 u, float* f) {
  f[0] = bf2f(u.x & 0xffffu); f[1] = bf2f(u.x >> 16);
  f[2] = bf2f(u.y & 0xffffu); f[3] = bf2f(u.y >> 16);
  f[4] = bf2f(u.z & 0xffffu); f[5] = bf2f(u.z >> 16);
  f[6] = bf2f(u.w & 0xffffu); f[7] = bf2f(u.w >> 16);
}

union AFrag { uint4 q[2]; v16bf v; };

// ---------------------------------------------------------------------------
// Generic bf16 GEMM: C[M,Nc] = A[M,K](bf16, lda=K) * Bpacked + bias, opt ReLU.
// B packed as [K/32][Nc][32] bf16 (column chunks contiguous in K).
// Block = 256 threads (8 waves); block tile 64x64; wave tile 16x32.
// Fragment layouts used here (contiguous k, same order for A and B):
//  A (16x32 bf16): lane<16 -> M=lane, K 0..15 ; lane>=16 -> M=lane-16, K 16..31
//  B (32x16 bf16): lane<16 -> N=lane, K 0..15 ; lane>=16 -> N=lane-16, K 16..31
//  C (16x16 f32):  vgpr r, lane l -> M = r + 8*(l>>4), N = l&15
// ---------------------------------------------------------------------------
template <int OUT_BF16, int RELU>
__global__ void __launch_bounds__(256)
gemm_bf16_kernel(const unsigned short* __restrict__ A,
                 const unsigned short* __restrict__ Bp,
                 void* __restrict__ Cout,
                 const float* __restrict__ bias,
                 int M, int K, int Nc, int ldc)
{
  const int wave = threadIdx.x >> 5;
  const int lane = threadIdx.x & 31;
  const int lh   = lane >> 4;       // half-id
  const int l15  = lane & 15;
  const int m0 = blockIdx.x * 64 + (wave & 3) * 16;
  const int n0 = blockIdx.y * 64 + (wave >> 2) * 32;

  int arow = m0 + l15; if (arow >= M) arow = M - 1;   // clamp, store is guarded
  const unsigned short* ap  = A  + (size_t)arow * K + lh * 16;
  const unsigned short* bp0 = Bp + (size_t)(n0 + l15) * 32 + lh * 16;
  const unsigned short* bp1 = bp0 + 16 * 32;

  v8f acc0 = {0.f,0.f,0.f,0.f,0.f,0.f,0.f,0.f};
  v8f acc1 = {0.f,0.f,0.f,0.f,0.f,0.f,0.f,0.f};

  const int ktiles = K >> 5;
  for (int kt = 0; kt < ktiles; ++kt) {
    AFrag af;
    af.q[0] = *(const uint4*)(ap);        // K: kt*32 + lh*16 .. +7
    af.q[1] = *(const uint4*)(ap + 8);    // K: kt*32 + lh*16 + 8 .. +15
    ap += 32;
    __builtin_prefetch(ap, 0, 3);
    const size_t boff = (size_t)kt * Nc * 32;
    v16bf bv0 = *(const v16bf*)(bp0 + boff);  // 32B contiguous, 32B aligned
    v16bf bv1 = *(const v16bf*)(bp1 + boff);
    acc0 = __builtin_amdgcn_wmma_f32_16x16x32_bf16(false, af.v, false, bv0,
                                                   (short)0, acc0, false, false);
    acc1 = __builtin_amdgcn_wmma_f32_16x16x32_bf16(false, af.v, false, bv1,
                                                   (short)0, acc1, false, false);
  }

  const int c0 = n0 + l15, c1 = c0 + 16;
  const float bia0 = bias ? bias[c0] : 0.f;
  const float bia1 = bias ? bias[c1] : 0.f;
#pragma unroll
  for (int r = 0; r < 8; ++r) {
    const int row = m0 + lh * 8 + r;
    if (row < M) {
      float v0 = acc0[r] + bia0;
      float v1 = acc1[r] + bia1;
      if (RELU) { v0 = fmaxf(v0, 0.f); v1 = fmaxf(v1, 0.f); }
      if (OUT_BF16) {
        unsigned short* o = (unsigned short*)Cout;
        o[(size_t)row * ldc + c0] = f2bf(v0);
        o[(size_t)row * ldc + c1] = f2bf(v1);
      } else {
        float* o = (float*)Cout;
        o[(size_t)row * ldc + c0] = v0;
        o[(size_t)row * ldc + c1] = v1;
      }
    }
  }
}

// ---- weight packing -------------------------------------------------------
// dst layout: [k/32][Nc][32] bf16
__global__ void pack_plain_kernel(unsigned short* __restrict__ dst,
                                  const float* __restrict__ src,
                                  int K, int Nc, int k0)
{
  int idx = blockIdx.x * blockDim.x + threadIdx.x;
  if (idx >= K * Nc) return;
  int k = idx / Nc, n = idx - k * Nc;
  int kk = k0 + k;
  dst[((size_t)(kk >> 5) * Nc + n) * 32 + (kk & 31)] = f2bf(src[(size_t)k * Nc + n]);
}

// W1[k, b*256+o] = bases[b][k][o];  W1[k, 2048+o] = root[k][o]   (Nc = 2304)
__global__ void pack_bases_kernel(unsigned short* __restrict__ dst,
                                  const float* __restrict__ bases,
                                  const float* __restrict__ root)
{
  int idx = blockIdx.x * blockDim.x + threadIdx.x;
  if (idx >= C_DIM * YW) return;
  int k = idx / YW, n = idx - k * YW;
  float v;
  if (n < NBAS * C_DIM) {
    int b = n >> 8, o = n & 255;
    v = bases[((size_t)b * C_DIM + k) * C_DIM + o];
  } else {
    v = root[(size_t)k * C_DIM + (n - NBAS * C_DIM)];
  }
  dst[((size_t)(k >> 5) * YW + n) * 32 + (k & 31)] = f2bf(v);
}

// ---- elementwise kernels --------------------------------------------------
__global__ void fuse_kernel(unsigned short* __restrict__ xb,
                            unsigned short* __restrict__ featb,
                            const float* __restrict__ enc,
                            const int* __restrict__ spk,
                            const float* __restrict__ table, int N)
{
  int idx = blockIdx.x * blockDim.x + threadIdx.x;
  if (idx >= N * C_DIM) return;
  int i = idx >> 8, k = idx & 255;
  float e = enc[idx];
  xb[idx] = f2bf(e + table[spk[i] * C_DIM + k]);
  featb[(size_t)i * 768 + k] = f2bf(e);          // feat[:,0:256] = encoding
}

__global__ void zero_kernel(unsigned int* __restrict__ p, size_t n)
{
  size_t i = (size_t)blockIdx.x * blockDim.x + threadIdx.x;
  size_t stride = (size_t)gridDim.x * blockDim.x;
  for (; i < n; i += stride) p[i] = 0u;
}

__global__ void count_kernel(int* __restrict__ cnt, const int* __restrict__ dst,
                             const int* __restrict__ et, int E)
{
  int e = blockIdx.x * blockDim.x + threadIdx.x;
  if (e >= E) return;
  atomicAdd(&cnt[dst[e] * NREL + et[e]], 1);
}

// h[i,k] = bias[k] + y[i, 2048+k]   (root contribution + bias)
__global__ void init_out_kernel(float* __restrict__ out,
                                const unsigned short* __restrict__ y,
                                const float* __restrict__ bias, int N)
{
  int idx = blockIdx.x * blockDim.x + threadIdx.x;
  if (idx >= N * C_DIM) return;
  int i = idx >> 8, k = idx & 255;
  out[idx] = bias[k] + bf2f((unsigned int)y[(size_t)i * YW + (NBAS * C_DIM) + k]);
}

// one wave per edge: msg = (1/cnt[dst,rel]) * sum_b comp[rel,b] * y[src, b*256 + :]
__global__ void __launch_bounds__(256)
rgcn_edge_kernel(float* __restrict__ out, const unsigned short* __restrict__ y,
                 const float* __restrict__ comp, const int* __restrict__ cnt,
                 const int* __restrict__ src, const int* __restrict__ dst,
                 const int* __restrict__ et, int E)
{
  int e = (int)(((size_t)blockIdx.x * blockDim.x + threadIdx.x) >> 5);
  int lane = threadIdx.x & 31;
  if (e >= E) return;
  int s = src[e], d = dst[e], r = et[e];
  float norm = 1.f / fmaxf((float)cnt[d * NREL + r], 1.f);
  float acc[8] = {0.f,0.f,0.f,0.f,0.f,0.f,0.f,0.f};
  const unsigned short* yb = y + (size_t)s * YW + lane * 8;
#pragma unroll
  for (int b = 0; b < NBAS; ++b) {
    float cb = comp[r * NBAS + b] * norm;
    uint4 u = *(const uint4*)(yb + b * C_DIM);
    float f[8]; unpack8(u, f);
#pragma unroll
    for (int j = 0; j < 8; ++j) acc[j] += cb * f[j];
  }
  float* op = out + (size_t)d * C_DIM + lane * 8;
#pragma unroll
  for (int j = 0; j < 8; ++j) atomicAdd(op + j, acc[j]);
}

// one wave per edge: agg[dst,:] += h[src,:]
__global__ void __launch_bounds__(256)
gconv_scatter_kernel(float* __restrict__ agg, const float* __restrict__ h,
                     const int* __restrict__ src, const int* __restrict__ dst, int E)
{
  int e = (int)(((size_t)blockIdx.x * blockDim.x + threadIdx.x) >> 5);
  int lane = threadIdx.x & 31;
  if (e >= E) return;
  int s = src[e], d = dst[e];
  const float* hp = h + (size_t)s * C_DIM + lane * 8;
  float* op = agg + (size_t)d * C_DIM + lane * 8;
  float4 a = *(const float4*)hp;
  float4 b = *(const float4*)(hp + 4);
  atomicAdd(op + 0, a.x); atomicAdd(op + 1, a.y);
  atomicAdd(op + 2, a.z); atomicAdd(op + 3, a.w);
  atomicAdd(op + 4, b.x); atomicAdd(op + 5, b.y);
  atomicAdd(op + 6, b.z); atomicAdd(op + 7, b.w);
}

// fp32 [N,256] -> bf16 with arbitrary destination row stride
__global__ void conv256_kernel(unsigned short* __restrict__ dst, int dstStride,
                               const float* __restrict__ src, int N)
{
  int idx = blockIdx.x * blockDim.x + threadIdx.x;
  if (idx >= N * C_DIM) return;
  int i = idx >> 8, k = idx & 255;
  dst[(size_t)i * dstStride + k] = f2bf(src[idx]);
}

// final 64 -> 1 projection
__global__ void final_dot_kernel(float* __restrict__ out, const float* __restrict__ m2,
                                 const float* __restrict__ w, const float* __restrict__ b,
                                 int N)
{
  int i = blockIdx.x * blockDim.x + threadIdx.x;
  if (i >= N) return;
  const float* mp = m2 + (size_t)i * 64;
  float acc = b[0];
#pragma unroll
  for (int j = 0; j < 64; ++j) acc += mp[j] * w[j];
  out[i] = acc;
}

// ---------------------------------------------------------------------------
extern "C" void kernel_launch(void* const* d_in, const int* in_sizes, int n_in,
                              void* d_out, int out_size, void* d_ws, size_t ws_size,
                              hipStream_t stream)
{
  (void)n_in; (void)out_size; (void)ws_size;
  const float* enc    = (const float*)d_in[0];
  const int*   spk    = (const int*)  d_in[1];
  const int*   eidx   = (const int*)  d_in[2];
  const int*   etype  = (const int*)  d_in[3];
  const int*   estype = (const int*)  d_in[4];
  const float* table  = (const float*)d_in[5];
  const float* comp1  = (const float*)d_in[6];
  const float* bases1 = (const float*)d_in[7];
  const float* root1  = (const float*)d_in[8];
  const float* b1     = (const float*)d_in[9];
  const float* w2rel  = (const float*)d_in[10];
  const float* b2     = (const float*)d_in[11];
  const float* w2root = (const float*)d_in[12];
  const float* comp3  = (const float*)d_in[13];
  const float* bases3 = (const float*)d_in[14];
  const float* root3  = (const float*)d_in[15];
  const float* b3     = (const float*)d_in[16];
  const float* w4rel  = (const float*)d_in[17];
  const float* b4     = (const float*)d_in[18];
  const float* w4root = (const float*)d_in[19];
  const float* wd1    = (const float*)d_in[20];
  const float* bd1    = (const float*)d_in[21];
  const float* wd2    = (const float*)d_in[22];
  const float* bd2    = (const float*)d_in[23];
  const float* wd3    = (const float*)d_in[24];
  const float* bd3    = (const float*)d_in[25];

  const int N = in_sizes[1];             // 60000
  const int E = in_sizes[3];             // 480000
  const int* srcI = eidx;
  const int* dstI = eidx + E;

  // ---- carve workspace (256B aligned slabs) ----
  char* w = (char*)d_ws;
  auto alloc = [&](size_t bytes) { char* p = w; w += (bytes + 255) & ~(size_t)255; return p; };
  unsigned short* xb    = (unsigned short*)alloc((size_t)N * 256 * 2);   // bf16 x
  unsigned short* featb = (unsigned short*)alloc((size_t)N * 768 * 2);   // MLP input
  unsigned short* ybuf  = (unsigned short*)alloc((size_t)N * YW  * 2);   // x @ [bases|root]
  float*          hbuf  = (float*)         alloc((size_t)N * 256 * 4);   // RGCN out (fp32)
  float*          agg   = (float*)         alloc((size_t)N * 256 * 4);   // gconv sum
  unsigned short* ab    = (unsigned short*)alloc((size_t)N * 512 * 2);   // [h | agg] bf16
  unsigned short* m1b   = (unsigned short*)alloc((size_t)N * 128 * 2);
  float*          m2    = (float*)         alloc((size_t)N * 64  * 4);
  int*            cnt   = (int*)           alloc((size_t)N * 16  * 4);
  unsigned short* B1p   = (unsigned short*)alloc((size_t)256 * YW * 2);
  unsigned short* B3p   = (unsigned short*)alloc((size_t)256 * YW * 2);
  unsigned short* Wg2p  = (unsigned short*)alloc((size_t)512 * 256 * 2);
  unsigned short* Wg4p  = (unsigned short*)alloc((size_t)512 * 256 * 2);
  unsigned short* wd1p  = (unsigned short*)alloc((size_t)768 * 128 * 2);
  unsigned short* wd2p  = (unsigned short*)alloc((size_t)128 * 64  * 2);

  const int T = 256;
  auto cdiv = [](int a, int b) { return (a + b - 1) / b; };

  // ---- pack weights into WMMA-friendly layout ----
  pack_bases_kernel<<<cdiv(256 * YW, T), T, 0, stream>>>(B1p, bases1, root1);
  pack_bases_kernel<<<cdiv(256 * YW, T), T, 0, stream>>>(B3p, bases3, root3);
  pack_plain_kernel<<<cdiv(256 * 256, T), T, 0, stream>>>(Wg2p, w2root, 256, 256, 0);
  pack_plain_kernel<<<cdiv(256 * 256, T), T, 0, stream>>>(Wg2p, w2rel,  256, 256, 256);
  pack_plain_kernel<<<cdiv(256 * 256, T), T, 0, stream>>>(Wg4p, w4root, 256, 256, 0);
  pack_plain_kernel<<<cdiv(256 * 256, T), T, 0, stream>>>(Wg4p, w4rel,  256, 256, 256);
  pack_plain_kernel<<<cdiv(768 * 128, T), T, 0, stream>>>(wd1p, wd1, 768, 128, 0);
  pack_plain_kernel<<<cdiv(128 * 64,  T), T, 0, stream>>>(wd2p, wd2, 128, 64, 0);

  // ---- x = encoding + spk_table[speaker] ; feat[:,0:256] = encoding ----
  fuse_kernel<<<cdiv(N * 256, T), T, 0, stream>>>(xb, featb, enc, spk, table, N);

  dim3 gy (cdiv(N, 64), YW / 64);
  dim3 gg (cdiv(N, 64), 256 / 64);
  dim3 gm1(cdiv(N, 64), 128 / 64);
  dim3 gm2(cdiv(N, 64), 64 / 64);
  const int edgeBlocks = cdiv(E, 8);   // 8 waves/block, 1 edge/wave

  for (int pass = 0; pass < 2; ++pass) {
    const unsigned short* Bp   = pass ? B3p    : B1p;
    const float*          comp = pass ? comp3  : comp1;
    const float*          bb   = pass ? b3     : b1;
    const int*            et   = pass ? estype : etype;
    const unsigned short* Wgp  = pass ? Wg4p   : Wg2p;
    const float*          gb   = pass ? b4     : b2;
    unsigned short*       gout = featb + (pass ? 512 : 256);

    // y = x @ [bases | root]  (bf16 out, [N,2304])
    gemm_bf16_kernel<1, 0><<<gy, T, 0, stream>>>(xb, Bp, ybuf, nullptr, N, 256, YW, YW);
    // per-(dst, rel) degree
    zero_kernel<<<2048, T, 0, stream>>>((unsigned int*)cnt, (size_t)N * 16);
    count_kernel<<<cdiv(E, T), T, 0, stream>>>(cnt, dstI, et, E);
    // h = bias + root part; then scatter basis-combined messages
    init_out_kernel<<<cdiv(N * 256, T), T, 0, stream>>>(hbuf, ybuf, bb, N);
    rgcn_edge_kernel<<<edgeBlocks, T, 0, stream>>>(hbuf, ybuf, comp, cnt, srcI, dstI, et, E);
    // GraphConv: agg = Σ h[src]; g = [h | agg] @ [w_root ; w_rel] + b
    zero_kernel<<<2048, T, 0, stream>>>((unsigned int*)agg, (size_t)N * 256);
    conv256_kernel<<<cdiv(N * 256, T), T, 0, stream>>>(ab, 512, hbuf, N);
    gconv_scatter_kernel<<<edgeBlocks, T, 0, stream>>>(agg, hbuf, srcI, dstI, E);
    conv256_kernel<<<cdiv(N * 256, T), T, 0, stream>>>(ab + 256, 512, agg, N);
    gemm_bf16_kernel<1, 0><<<gg, T, 0, stream>>>(ab, Wgp, gout, gb, N, 512, 256, 768);
  }

  // ---- MLP on feat = [encoding | h1 | h2] ----
  gemm_bf16_kernel<1, 1><<<gm1, T, 0, stream>>>(featb, wd1p, m1b, bd1, N, 768, 128, 128);
  gemm_bf16_kernel<0, 1><<<gm2, T, 0, stream>>>(m1b, wd2p, m2, bd2, N, 128, 64, 64);
  final_dot_kernel<<<cdiv(N, T), T, 0, stream>>>((float*)d_out, m2, wd3, bd3, N);
}
